// PipelinedMoELayer_28518582845652
// MI455X (gfx1250) — matrix operand order repaired and checked by reading, not compiled, not run on hardware
//
#include <hip/hip_runtime.h>
#include <hip/hip_bf16.h>

#define NUM_EXPERTS 8
#define D_MODEL     1024
#define D_OUT       1024
#define T_TOKENS    4096

typedef __attribute__((ext_vector_type(16))) __bf16 v16bf;
typedef __attribute__((ext_vector_type(8)))  __bf16 v8bf;
typedef __attribute__((ext_vector_type(8)))  float  v8f;

union V16 { v16bf v; v8bf h[2]; __bf16 e[16]; };
union V8  { v8f   v; float  e[8];  };
union V8BF { v8bf v; __bf16 e[8]; };

// ---------------------------------------------------------------------------
// Kernel 0: zero per-expert token counters (ws is poisoned once, never
// re-poisoned between timed replays, so zero on every call).
// ---------------------------------------------------------------------------
__global__ void moe_zero_counts(int* __restrict__ counts) {
    if (threadIdx.x < NUM_EXPERTS) counts[threadIdx.x] = 0;
}

// ---------------------------------------------------------------------------
// Kernel 1: gating + top-1 routing. One wave32 per token.
// softmax is monotone -> argmax(logits) == argmax(softmax(logits)).
// Tie-break: ascending e with strict '>' == jnp.argmax (first max wins).
// ---------------------------------------------------------------------------
__global__ __launch_bounds__(256)
void moe_gate_route(const float* __restrict__ x,
                    const float* __restrict__ gate_w,
                    const float* __restrict__ gate_b,
                    int* __restrict__ counts,
                    int* __restrict__ tok_list) {
    const int wave = (blockIdx.x * blockDim.x + threadIdx.x) >> 5;
    const int lane = threadIdx.x & 31;
    if (wave >= T_TOKENS) return;

    const float* xr = x + (size_t)wave * D_MODEL;
    float acc[NUM_EXPERTS];
#pragma unroll
    for (int e = 0; e < NUM_EXPERTS; ++e) acc[e] = 0.0f;

    for (int d = lane; d < D_MODEL; d += 32) {
        const float xv = xr[d];
        const float* g = gate_w + (size_t)d * NUM_EXPERTS;
#pragma unroll
        for (int e = 0; e < NUM_EXPERTS; ++e)
            acc[e] = fmaf(xv, g[e], acc[e]);
    }
#pragma unroll
    for (int e = 0; e < NUM_EXPERTS; ++e) {
#pragma unroll
        for (int off = 16; off >= 1; off >>= 1)
            acc[e] += __shfl_xor(acc[e], off, 32);
    }
    if (lane == 0) {
        int best = 0;
        float bv = acc[0] + gate_b[0];
#pragma unroll
        for (int e = 1; e < NUM_EXPERTS; ++e) {
            const float v = acc[e] + gate_b[e];
            if (v > bv) { bv = v; best = e; }
        }
        const int slot = atomicAdd(&counts[best], 1);
        tok_list[best * T_TOKENS + slot] = wave;
    }
}

// ---------------------------------------------------------------------------
// Kernel 2: grouped expert GEMM, split-bf16 WMMA (fp32-class accuracy:
// x*w ~= xh*wh + xh*wl + xl*wh, residual ~2^-18 relative).
// Block = 256 threads = 8 waves. Block tile: M=64 routed tokens x N=128.
// - A is gathered, converted to bf16 hi/lo ONCE per block during staging and
//   kept in two LDS planes -> fragment build is pure ds_load_b128 (this kernel's
//   16-bit A order = contiguous K, 8-wide runs at offsets {kb, kb+8}).
// - All 16 fragment loads for a slab are issued BEFORE the 12-WMMA chain so
//   LDS latency drains under the matrix ops (no dscnt==0 stall per WMMA).
// - LDS is double-buffered; raw global loads for slab i+1 are issued before
//   the WMMAs of slab i; ONE barrier per slab.
// - B (L2-resident expert_w columns) converted per wave: its columns are
//   partitioned across waves, so there is no intra-block redundancy.
// ---------------------------------------------------------------------------
__global__ __launch_bounds__(256)
void moe_expert_gemm(const float* __restrict__ x,
                     const float* __restrict__ expert_w,
                     const float* __restrict__ expert_b,
                     const int*   __restrict__ counts,
                     const int*   __restrict__ tok_list,
                     float*       __restrict__ out) {
    const int e   = blockIdx.z;
    const int cnt = counts[e];
    const int m0  = blockIdx.y * 64;
    if (m0 >= cnt) return;                 // uniform per block: EXEC stays all-1s
    const int n0  = blockIdx.x * 128;

    const int tid   = threadIdx.x;
    const int wave  = tid >> 5;
    const int lane  = tid & 31;
    const int lhalf = lane >> 4;           // 0: lanes 0-15, 1: lanes 16-31
    const int l15   = lane & 15;
    const int kb    = lhalf * 16;          // A-fragment K base (contiguous K, matches B)
    const int kB    = lhalf * 16;          // B-fragment K base (contiguous K, matches A)

    // bf16 hi/lo planes, double buffered. Row stride 48 bf16 = 96 B (16B mult).
    __shared__ __bf16 AhS[2][64][48];
    __shared__ __bf16 AlS[2][64][48];

    const int* tl = tok_list + e * T_TOKENS;

    // ---- staging assignment: 4 threads per token row, hoisted out of K loop
    const int srow  = tid >> 2;            // 0..63
    const int sq8   = (tid & 3) * 8;       // 8-float chunk within the 32-col slab
    const int smrow = m0 + srow;
    const int stok  = tl[smrow < cnt ? smrow : (cnt - 1)];   // clamp; masked at store
    const float* sxrow = x + (size_t)stok * D_MODEL + sq8;

    // ---- this lane's fixed output column
    const int n = n0 + wave * 16 + l15;
    const float* wcol = expert_w + (size_t)e * D_MODEL * D_OUT + n;
    const float bn = expert_b[e * D_OUT + n];

    // convert 8 floats -> hi/lo bf16 and store to LDS planes
    auto stageStore = [&](int buf, const float4& a0, const float4& a1) {
        const float f[8] = {a0.x, a0.y, a0.z, a0.w, a1.x, a1.y, a1.z, a1.w};
        V8BF h, l;
#pragma unroll
        for (int i = 0; i < 8; ++i) {
            const __bf16 hb = (__bf16)f[i];
            h.e[i] = hb;
            l.e[i] = (__bf16)(f[i] - (float)hb);
        }
        *(v8bf*)&AhS[buf][srow][sq8] = h.v;
        *(v8bf*)&AlS[buf][srow][sq8] = l.v;
    };

    V8 acc[4];
#pragma unroll
    for (int s = 0; s < 4; ++s)
#pragma unroll
        for (int i = 0; i < 8; ++i) acc[s].e[i] = 0.0f;

    // ---- prologue: stage slab 0, prefetch B raw for slab 0
    {
        const float4 a0 = *(const float4*)(sxrow);
        const float4 a1 = *(const float4*)(sxrow + 4);
        stageStore(0, a0, a1);
    }
    float bRaw[16];
#pragma unroll
    for (int i = 0; i < 16; ++i) bRaw[i] = wcol[(size_t)(kB + i) * D_OUT];
    __syncthreads();

    int p = 0;
    for (int kc = 0; kc < D_MODEL; kc += 32) {
        const bool hasNext = (kc + 32 < D_MODEL);     // uniform
        // ---- issue raw global loads for slab kc+32 (hide behind WMMAs) ----
        float4 a0, a1;
        if (hasNext) {
            a0 = *(const float4*)(sxrow + kc + 32);
            a1 = *(const float4*)(sxrow + kc + 36);
        }
        // ---- batch ALL A-fragment LDS loads for this slab up front ----
        V16 ah[4], al[4];
#pragma unroll
        for (int s = 0; s < 4; ++s) {
            const int m = s * 16 + l15;
            ah[s].h[0] = *(const v8bf*)&AhS[p][m][kb];
            ah[s].h[1] = *(const v8bf*)&AhS[p][m][kb + 8];
            al[s].h[0] = *(const v8bf*)&AlS[p][m][kb];
            al[s].h[1] = *(const v8bf*)&AlS[p][m][kb + 8];
        }
        // ---- convert current B fragments (overlaps LDS drain) ----
        V16 bh, bl;
#pragma unroll
        for (int i = 0; i < 16; ++i) {
            const __bf16 hb = (__bf16)bRaw[i];
            bh.e[i] = hb;
            bl.e[i] = (__bf16)(bRaw[i] - (float)hb);
        }
        // ---- prefetch next B raw (strided dwords, L2 hits) ----
        if (hasNext) {
#pragma unroll
            for (int i = 0; i < 16; ++i)
                bRaw[i] = wcol[(size_t)(kc + 32 + kB + i) * D_OUT];
        }
        // ---- 12 WMMAs: 4 independent accumulator chains ----
#pragma unroll
        for (int s = 0; s < 4; ++s) {
            acc[s].v = __builtin_amdgcn_wmma_f32_16x16x32_bf16(
                false, ah[s].v, false, bh.v, (short)0, acc[s].v, false, false);
            acc[s].v = __builtin_amdgcn_wmma_f32_16x16x32_bf16(
                false, ah[s].v, false, bl.v, (short)0, acc[s].v, false, false);
            acc[s].v = __builtin_amdgcn_wmma_f32_16x16x32_bf16(
                false, al[s].v, false, bh.v, (short)0, acc[s].v, false, false);
        }
        // ---- stage slab kc+32 into the other buffer; ONE barrier per slab ----
        if (hasNext) stageStore(1 - p, a0, a1);
        __syncthreads();
        p ^= 1;
    }

    // ---- bias add + scatter store (C/D layout: VGPR r -> M = r + 8*lhalf) ----
    const bool full = (m0 + 64 <= cnt);    // uniform fast path for interior blocks
    if (full) {
#pragma unroll
        for (int s = 0; s < 4; ++s)
#pragma unroll
            for (int r = 0; r < 8; ++r) {
                const int mrow = m0 + s * 16 + r + lhalf * 8;
                out[(size_t)tl[mrow] * D_OUT + n] = acc[s].e[r] + bn;
            }
    } else {
#pragma unroll
        for (int s = 0; s < 4; ++s)
#pragma unroll
            for (int r = 0; r < 8; ++r) {
                const int mrow = m0 + s * 16 + r + lhalf * 8;
                if (mrow < cnt)
                    out[(size_t)tl[mrow] * D_OUT + n] = acc[s].e[r] + bn;
            }
    }
}

// ---------------------------------------------------------------------------
extern "C" void kernel_launch(void* const* d_in, const int* in_sizes, int n_in,
                              void* d_out, int out_size, void* d_ws, size_t ws_size,
                              hipStream_t stream) {
    (void)in_sizes; (void)n_in; (void)out_size; (void)ws_size;

    const float* x        = (const float*)d_in[0];   // [2,2048,1024]
    const float* gate_w   = (const float*)d_in[1];   // [1024,8]
    const float* gate_b   = (const float*)d_in[2];   // [8]
    const float* expert_w = (const float*)d_in[3];   // [8,1024,1024]
    const float* expert_b = (const float*)d_in[4];   // [8,1024]
    float* out = (float*)d_out;                      // [2,2048,1024]

    int* counts   = (int*)d_ws;                      // 8 ints (256 B slot)
    int* tok_list = (int*)((char*)d_ws + 256);       // 8 * 4096 ints

    moe_zero_counts<<<1, 64, 0, stream>>>(counts);

    // one wave32 per token: 4096 waves = 512 blocks x 256 threads
    moe_gate_route<<<T_TOKENS / 8, 256, 0, stream>>>(x, gate_w, gate_b,
                                                     counts, tok_list);

    // grid: (N tiles, worst-case M/64 tiles, experts); inactive blocks exit fast
    dim3 grid(D_OUT / 128, T_TOKENS / 64, NUM_EXPERTS);
    moe_expert_gemm<<<grid, 256, 0, stream>>>(x, expert_w, expert_b,
                                              counts, tok_list, out);
}
